// PINN_NDE_V4_79714593013862
// MI455X (gfx1250) — matrix operand order repaired and checked by reading, not compiled, not run on hardware
//
#include <hip/hip_runtime.h>
#include <hip/hip_bf16.h>
#include <math.h>

typedef __attribute__((ext_vector_type(16))) _Float16 v16h;
typedef __attribute__((ext_vector_type(8)))  _Float16 h8;
typedef __attribute__((ext_vector_type(8)))  float    v8f;
typedef __attribute__((ext_vector_type(4)))  float    f4;

#define ACT_NONE 0
#define ACT_GELU 1
#define ACT_SIGMOID 2

__device__ __forceinline__ float gelu_f(float x) {
    return 0.5f * x * (1.0f + erff(x * 0.70710678118654752440f));
}
__device__ __forceinline__ float sigm_f(float x) {
    return 1.0f / (1.0f + expf(-x));
}
__device__ __forceinline__ float act_apply(float v, int act) {
    if (act == ACT_GELU) return gelu_f(v);
    if (act == ACT_SIGMOID) return sigm_f(v);
    return v;
}

// ---------------------------------------------------------------------------
// Fragment loaders (wave32 layouts per CDNA5 ISA).
//   A frag (16x32, 16-bit): lane m = lane%16; hsel = lane>=16
//       a[0..7]  <- k0 + 8*hsel + i
//       a[8..15] <- k0 + 16 + 8*hsel + i
//   B frag (32x16): lane n = lane%16; same k order as A frag  (row n of W[N,K])
// K is always a multiple of 32 and rows are 16B aligned -> unguarded vector loads.
// ---------------------------------------------------------------------------
__device__ __forceinline__ v16h load_a_f32(const float* __restrict__ ar, int k0, int hsel)
{
    const f4* p = (const f4*)(ar + k0 + hsel * 8);
    const f4* q = (const f4*)(ar + k0 + 16 + hsel * 8);
    f4 x0 = p[0], x1 = p[1], x2 = q[0], x3 = q[1];
    v16h a;
    #pragma unroll
    for (int i = 0; i < 4; i++) {
        a[i]      = (_Float16)x0[i];
        a[4 + i]  = (_Float16)x1[i];
        a[8 + i]  = (_Float16)x2[i];
        a[12 + i] = (_Float16)x3[i];
    }
    return a;
}
__device__ __forceinline__ v16h load_a_f16(const _Float16* __restrict__ ar, int k0, int hsel)
{
    const h8* p = (const h8*)(ar + k0 + hsel * 8);
    const h8* q = (const h8*)(ar + k0 + 16 + hsel * 8);
    h8 x0 = p[0], x1 = q[0];
    v16h a;
    #pragma unroll
    for (int i = 0; i < 8; i++) { a[i] = x0[i]; a[8 + i] = x1[i]; }
    return a;
}
__device__ __forceinline__ v16h load_b_f32(const float* __restrict__ wr, int k0, int hsel)
{
    const f4* p = (const f4*)(wr + k0 + hsel * 8);
    const f4* q = (const f4*)(wr + k0 + 16 + hsel * 8);
    f4 x0 = p[0], x1 = p[1], x2 = q[0], x3 = q[1];
    v16h b;
    #pragma unroll
    for (int i = 0; i < 4; i++) {
        b[i]      = (_Float16)x0[i];
        b[4 + i]  = (_Float16)x1[i];
        b[8 + i]  = (_Float16)x2[i];
        b[12 + i] = (_Float16)x3[i];
    }
    return b;
}

// ---------------------------------------------------------------------------
// WMMA GEMM:  C[M,N] = act( A[M,K] * W[N,K]^T + bias )
// One wave computes a 32x32 output tile (2x2 of 16x16), K in chunks of 32 via
// v_wmma_f32_16x16x32_f16.  Requires: M%32==0, N%32==0, K%32==0, rows 16B aligned.
// ---------------------------------------------------------------------------
template <bool A_F16>
__global__ __launch_bounds__(128)
void wmma_gemm(const void* __restrict__ Av, long lda,
               const float* __restrict__ Wt,   // [N,K] row-major
               const float* __restrict__ bias, // [N] or nullptr
               float* __restrict__ C, long ldc,
               int M, int N, int K, int act)
{
    int wave = threadIdx.x >> 5;
    int lane = threadIdx.x & 31;
    int tn32 = N >> 5;
    int tm32 = M >> 5;
    int tile = blockIdx.x * 4 + wave;
    if (tile >= tm32 * tn32) return;
    int tm = (tile / tn32) << 5;
    int tn = (tile - (tile / tn32) * tn32) << 5;
    int hsel = lane >> 4;
    int l16  = lane & 15;
    int m0 = tm + l16, m1 = m0 + 16;
    int n0 = tn + l16, n1 = n0 + 16;
    const float* w0 = Wt + (long)n0 * K;
    const float* w1 = Wt + (long)n1 * K;
    const float*    ar0f = ((const float*)Av)    + (long)m0 * lda;
    const float*    ar1f = ((const float*)Av)    + (long)m1 * lda;
    const _Float16* ar0h = ((const _Float16*)Av) + (long)m0 * lda;
    const _Float16* ar1h = ((const _Float16*)Av) + (long)m1 * lda;

    v8f acc00 = {}, acc01 = {}, acc10 = {}, acc11 = {};
    for (int k0 = 0; k0 < K; k0 += 32) {
        v16h a0 = A_F16 ? load_a_f16(ar0h, k0, hsel) : load_a_f32(ar0f, k0, hsel);
        v16h a1 = A_F16 ? load_a_f16(ar1h, k0, hsel) : load_a_f32(ar1f, k0, hsel);
        v16h b0 = load_b_f32(w0, k0, hsel);
        v16h b1 = load_b_f32(w1, k0, hsel);
        acc00 = __builtin_amdgcn_wmma_f32_16x16x32_f16(false, a0, false, b0, (short)0, acc00, false, false);
        acc01 = __builtin_amdgcn_wmma_f32_16x16x32_f16(false, a0, false, b1, (short)0, acc01, false, false);
        acc10 = __builtin_amdgcn_wmma_f32_16x16x32_f16(false, a1, false, b0, (short)0, acc10, false, false);
        acc11 = __builtin_amdgcn_wmma_f32_16x16x32_f16(false, a1, false, b1, (short)0, acc11, false, false);
    }
    float b0v = bias ? bias[n0] : 0.0f;
    float b1v = bias ? bias[n1] : 0.0f;
    #pragma unroll
    for (int r = 0; r < 8; r++) {
        int mo0 = tm + r + 8 * hsel;   // C frag: m = r + 8*(lane>=16)
        int mo1 = mo0 + 16;
        C[(long)mo0 * ldc + n0] = act_apply(acc00[r] + b0v, act);
        C[(long)mo0 * ldc + n1] = act_apply(acc01[r] + b1v, act);
        C[(long)mo1 * ldc + n0] = act_apply(acc10[r] + b0v, act);
        C[(long)mo1 * ldc + n1] = act_apply(acc11[r] + b1v, act);
    }
}

// ---------------------------------------------------------------------------
// Row LayerNorm: one wave per row. res_mode: 0 none, 1 pre-LN add, 2 post-LN add.
// ---------------------------------------------------------------------------
__global__ __launch_bounds__(32)
void ln_kernel(const float* __restrict__ src,
               const float* __restrict__ resid, int res_mode,
               const float* __restrict__ g, const float* __restrict__ bt,
               float* __restrict__ dst, int F, int act)
{
    long row = blockIdx.x;
    int lane = threadIdx.x;
    float xa[8];
    float s = 0.0f;
    int cnt = 0;
    for (int k = lane; k < F; k += 32) {
        float v = src[row * F + k];
        if (res_mode == 1) v += resid[row * F + k];
        xa[cnt++] = v;
        s += v;
    }
    for (int o = 16; o > 0; o >>= 1) s += __shfl_xor(s, o, 32);
    float mean = s / (float)F;
    float v2 = 0.0f;
    for (int i = 0; i < cnt; i++) { float d = xa[i] - mean; v2 += d * d; }
    for (int o = 16; o > 0; o >>= 1) v2 += __shfl_xor(v2, o, 32);
    float inv = rsqrtf(v2 / (float)F + 1e-5f);
    cnt = 0;
    for (int k = lane; k < F; k += 32) {
        float y = (xa[cnt] - mean) * inv * g[k] + bt[k];
        if (res_mode == 2) y += resid[row * F + k];
        if (act == ACT_GELU) y = gelu_f(y);
        dst[row * F + k] = y;
        cnt++;
    }
}

// im2col for Conv1d(k=5,pad=2) over token-major [B,T,C] -> f16 [B*T, C*5]
__global__ void im2col_k5(const float* __restrict__ x, _Float16* __restrict__ col,
                          int Bn, int T, int Cch)
{
    long idx = (long)blockIdx.x * blockDim.x + threadIdx.x;
    long tot = (long)Bn * T * Cch * 5;
    if (idx >= tot) return;
    int dt = (int)(idx % 5);
    long r = idx / 5;
    int ci = (int)(r % Cch);
    long bt = r / Cch;
    int t = (int)(bt % T);
    long b = bt / T;
    int ts = t + dt - 2;
    float v = (ts >= 0 && ts < T) ? x[(b * T + ts) * Cch + ci] : 0.0f;
    col[(b * T + t) * (long)(Cch * 5) + ci * 5 + dt] = (_Float16)v;
}

__global__ void bn_affine(float* __restrict__ y, const float* __restrict__ g,
                          const float* __restrict__ b, const float* __restrict__ m,
                          const float* __restrict__ v, int Cch, long tot)
{
    long idx = (long)blockIdx.x * blockDim.x + threadIdx.x;
    if (idx >= tot) return;
    int c = (int)(idx % Cch);
    y[idx] = (y[idx] - m[c]) * rsqrtf(v[c] + 1e-5f) * g[c] + b[c];
}

__global__ void add_kernel(const float* __restrict__ a, const float* __restrict__ b,
                           float* __restrict__ c, long n)
{
    long i = (long)blockIdx.x * blockDim.x + threadIdx.x;
    if (i < n) c[i] = a[i] + b[i];
}

__global__ void zero_kernel(float* __restrict__ p, long n)
{
    long i = (long)blockIdx.x * blockDim.x + threadIdx.x;
    if (i < n) p[i] = 0.0f;
}

// GRU gate fuse; grid.y = dir (0 fwd, 1 bwd)
__global__ void gru_gate(const float* __restrict__ xp_f, const float* __restrict__ xp_b,
                         const float* __restrict__ hp_f, const float* __restrict__ hp_b,
                         const float* __restrict__ hv_f, const float* __restrict__ hv_b,
                         float* __restrict__ hn_f, float* __restrict__ hn_b,
                         float* __restrict__ ys, int step, int T, int outF)
{
    int dir = blockIdx.y;
    long idx = (long)blockIdx.x * blockDim.x + threadIdx.x;
    if (idx >= 512L * 128) return;
    int j = (int)(idx % 128);
    long bb = idx / 128;
    int tidx = dir ? (T - 1 - step) : step;
    const float* xp = dir ? xp_b : xp_f;
    const float* hp = dir ? hp_b : hp_f;
    const float* hv = dir ? hv_b : hv_f;
    long xo = (bb * T + tidx) * 384;
    float xr = xp[xo + j], xz = xp[xo + 128 + j], xn = xp[xo + 256 + j];
    long ho = bb * 384;
    float hr = hp[ho + j], hz = hp[ho + 128 + j], hn = hp[ho + 256 + j];
    float r  = sigm_f(xr + hr);
    float zg = sigm_f(xz + hz);
    float n  = tanhf(xn + r * hn);
    float h  = (1.0f - zg) * n + zg * hv[bb * 128 + j];
    (dir ? hn_b : hn_f)[bb * 128 + j] = h;
    ys[(bb * T + tidx) * (long)outF + dir * 128 + j] = h;
}

// Attention: one block per (b,head); T=40, head_d=40
__global__ __launch_bounds__(256)
void attn_kernel(const float* __restrict__ qkv, float* __restrict__ out)
{
    __shared__ float Ks[1600], Vs[1600], As[8][40];
    int bh = blockIdx.x;
    int b = bh >> 2, h = bh & 3;
    const float* base = qkv + (long)b * 40 * 480;
    for (int i = threadIdx.x; i < 1600; i += 256) {
        int s = i / 40, d = i % 40;
        Ks[i] = base[s * 480 + 160 + h * 40 + d];
        Vs[i] = base[s * 480 + 320 + h * 40 + d];
    }
    __syncthreads();
    int wave = threadIdx.x >> 5, lane = threadIdx.x & 31;
    float scale = rsqrtf(40.0f);
    for (int row = wave; row < 40; row += 8) {   // uniform: 5 trips per wave
        const float* q = base + row * 480 + h * 40;
        float s0 = 0.0f, s1 = 0.0f;
        for (int d = 0; d < 40; d++) {
            float qd = q[d];
            s0 += qd * Ks[lane * 40 + d];
            if (lane < 8) s1 += qd * Ks[(lane + 32) * 40 + d];
        }
        s0 *= scale;
        s1 = (lane < 8) ? s1 * scale : -1e30f;
        float mx = fmaxf(s0, s1);
        for (int o = 16; o > 0; o >>= 1) mx = fmaxf(mx, __shfl_xor(mx, o, 32));
        float e0 = expf(s0 - mx);
        float e1 = (lane < 8) ? expf(s1 - mx) : 0.0f;
        float sum = e0 + e1;
        for (int o = 16; o > 0; o >>= 1) sum += __shfl_xor(sum, o, 32);
        float invs = 1.0f / sum;
        As[wave][lane] = e0 * invs;
        if (lane < 8) As[wave][lane + 32] = e1 * invs;
        __syncthreads();
        float o0 = 0.0f, o1 = 0.0f;
        for (int j = 0; j < 40; j++) {
            float aj = As[wave][j];
            o0 += aj * Vs[j * 40 + lane];
            if (lane < 8) o1 += aj * Vs[j * 40 + lane + 32];
        }
        float* orow = out + ((long)b * 40 + row) * 160 + h * 40;
        orow[lane] = o0;
        if (lane < 8) orow[lane + 32] = o1;
        __syncthreads();
    }
}

// Small output head: out[row, n] = act( dot(H[row,:64], W[n,:64]) + b[n] )
__global__ __launch_bounds__(128)
void head_out(const float* __restrict__ Hm, const float* __restrict__ Wn,
              const float* __restrict__ bn, float* __restrict__ outp,
              long ldo, int nout, int act, long Mr)
{
    long row = blockIdx.x * 4 + (threadIdx.x >> 5);
    if (row >= Mr) return;
    int lane = threadIdx.x & 31;
    const float* hr = Hm + row * 64;
    float h0 = hr[lane], h1 = hr[lane + 32];
    for (int n = 0; n < nout; n++) {
        float p = h0 * Wn[n * 64 + lane] + h1 * Wn[n * 64 + lane + 32];
        for (int o = 16; o > 0; o >>= 1) p += __shfl_xor(p, o, 32);
        if (lane == 0) outp[row * ldo + n] = act_apply(p + bn[n], act);
    }
}

__global__ void tmean_kernel(const float* __restrict__ ts, float* __restrict__ tm)
{
    int t = threadIdx.x;
    if (t >= 40) return;
    float s = 0.0f;
    for (int b = 0; b < 512; b++) s += ts[b * 40 + t];
    tm[t] = s / 512.0f;
}

// zw padded to stride 192: [z(160) | W(1) | zeros(31)]
__global__ void zw_build(const float* __restrict__ z, const float* __restrict__ w,
                         float* __restrict__ zw, long M)
{
    long i = (long)blockIdx.x * blockDim.x + threadIdx.x;
    if (i >= M * 192) return;
    long m = i / 192;
    int k = (int)(i % 192);
    zw[i] = (k < 160) ? z[m * 160 + k] : ((k == 160) ? w[m] : 0.0f);
}

// pad [64,161] weights to [64,192] (zeros beyond 161)
__global__ void pad_w64(const float* __restrict__ w, float* __restrict__ out)
{
    int i = blockIdx.x * blockDim.x + threadIdx.x;
    if (i >= 64 * 192) return;
    int n = i / 192, k = i % 192;
    out[i] = (k < 161) ? w[n * 161 + k] : 0.0f;
}

// ---------------------------------------------------------------------------
// Neural-ODE RK2 integrator: one 160-thread block per batch row.
// ---------------------------------------------------------------------------
__device__ float ode_rate_blk(float Wv, float tt, float sev, int tid,
                              const float* zproj_s, float wcol, float tcol,
                              float* hh, float* h2, float* red,
                              const float* nlg, const float* nlb,
                              const float* n2w, const float* n2b,
                              const float* n3w, float n3b0,
                              float ea, float eb, float ec, float ed)
{
    float h = zproj_s[tid] + Wv * wcol + tt * tcol;
    hh[tid] = h; __syncthreads();
    if (tid == 0) { float s = 0.0f; for (int k = 0; k < 160; k++) s += hh[k]; red[0] = s / 160.0f; }
    __syncthreads();
    float mean = red[0];
    float dv = h - mean;
    hh[tid] = dv * dv; __syncthreads();
    if (tid == 0) { float s = 0.0f; for (int k = 0; k < 160; k++) s += hh[k]; red[0] = rsqrtf(s / 160.0f + 1e-5f); }
    __syncthreads();
    float inv = red[0];
    float g1 = gelu_f((h - mean) * inv * nlg[tid] + nlb[tid]);
    hh[tid] = g1; __syncthreads();
    float s2 = n2b[tid];
    const float* w2 = n2w + (long)tid * 160;
    for (int k = 0; k < 160; k++) s2 += hh[k] * w2[k];
    h2[tid] = gelu_f(s2); __syncthreads();
    if (tid == 0) { float s = n3b0; for (int k = 0; k < 160; k++) s += h2[k] * n3w[k]; red[0] = s; }
    __syncthreads();
    float corr = red[0];
    float phys = ea * expf(eb * Wv) * (1.0f + ec * tt + ed * sev);
    float pre = 2.5f * (phys + corr);
    float sp = (pre > 20.0f) ? pre : log1pf(expf(pre));
    __syncthreads();  // allow hh reuse next call
    return sp / 2.5f + 1e-6f;
}

__global__ __launch_bounds__(160)
void ode_kernel(const float* __restrict__ ZF, const float* __restrict__ tm,
                const float* la, const float* lb, const float* lc, const float* ld,
                const float* __restrict__ sev1w, const float* __restrict__ sev1b,
                const float* __restrict__ sev2w, const float* __restrict__ sev2b,
                const float* __restrict__ n1w, const float* __restrict__ n1b,
                const float* __restrict__ nlg, const float* __restrict__ nlb,
                const float* __restrict__ n2w, const float* __restrict__ n2b,
                const float* __restrict__ n3w, const float* __restrict__ n3b,
                float* __restrict__ Wall)
{
    int b = blockIdx.x;
    int tid = threadIdx.x;
    __shared__ float zi[160], zproj[160], hh[160], h2[160], red[1], sevs[1], hsev[80];
    float ea = expf(la[0]), eb = expf(lb[0]), ec = expf(lc[0]), ed = expf(ld[0]);
    float n3b0 = n3b[0];
    float Wc = Wall[b * 40 + 0];
    for (int i = 0; i < 39; i++) {
        zi[tid] = ZF[((long)b * 40 + i + 1) * 160 + tid];
        __syncthreads();
        if (tid < 80) {
            float s = sev1b[tid];
            const float* wr = sev1w + (long)tid * 160;
            for (int k = 0; k < 160; k++) s += zi[k] * wr[k];
            hsev[tid] = gelu_f(s);
        }
        __syncthreads();
        if (tid == 0) {
            float s = sev2b[0];
            for (int k = 0; k < 80; k++) s += hsev[k] * sev2w[k];
            sevs[0] = sigm_f(s);
        }
        {
            float s = n1b[tid];
            const float* wr = n1w + (long)tid * 162;
            for (int k = 0; k < 160; k++) s += zi[k] * wr[k];
            zproj[tid] = s;
        }
        __syncthreads();
        float sev = sevs[0];
        float wcol = n1w[(long)tid * 162 + 160];
        float tcol = n1w[(long)tid * 162 + 161];
        float t0 = tm[i];
        float dt = (tm[i + 1] - tm[i]) / 8.0f;
        for (int j = 0; j < 8; j++) {
            float tt = t0 + j * dt;
            float k1 = ode_rate_blk(Wc, tt, sev, tid, zproj, wcol, tcol,
                                    hh, h2, red, nlg, nlb, n2w, n2b, n3w, n3b0,
                                    ea, eb, ec, ed);
            float k2 = ode_rate_blk(Wc + 0.5f * dt * k1, tt + 0.5f * dt, sev, tid,
                                    zproj, wcol, tcol, hh, h2, red,
                                    nlg, nlb, n2w, n2b, n3w, n3b0, ea, eb, ec, ed);
            Wc = Wc + dt * k2;
        }
        if (tid == 0) Wall[b * 40 + i + 1] = Wc;
        __syncthreads();
    }
}

// Final output assembly: [B*T, 171]
__global__ __launch_bounds__(192)
void assemble(const float* __restrict__ ZF, const float* __restrict__ Wall,
              const float* __restrict__ FL, const float* __restrict__ RU,
              const float* __restrict__ PH, const float* __restrict__ UN,
              const float* __restrict__ PY, float* __restrict__ out)
{
    long m = blockIdx.x;
    int c = threadIdx.x;
    if (c >= 171) return;
    float wear = Wall[m];
    float v;
    if (c == 0) v = wear;
    else if (c <= 3) {
        float f0 = FL[m * 3], f1 = FL[m * 3 + 1], f2 = FL[m * 3 + 2];
        float mx = fmaxf(f0, fmaxf(f1, f2));
        float e0 = expf(f0 - mx), e1 = expf(f1 - mx), e2 = expf(f2 - mx);
        float es = e0 + e1 + e2;
        float fe = (c == 1) ? e0 : ((c == 2) ? e1 : e2);
        v = 3.0f * wear * fe / es;
    }
    else if (c == 4)  v = RU[m];
    else if (c <= 7)  v = PH[m * 3 + (c - 5)];
    else if (c == 8)  v = fminf(2.0f, fmaxf(-6.0f, UN[m]));
    else if (c <= 10) { float x = PY[m * 2 + (c - 9)]; v = (x > 20.0f) ? x : log1pf(expf(x)); }
    else              v = ZF[m * 160 + (c - 11)];
    out[m * 171 + c] = v;
}

// ---------------------------------------------------------------------------
// Host side
// ---------------------------------------------------------------------------
static inline void launch_gemm_f32(const float* A, long lda, const float* W,
                                   const float* b, float* C, long ldc,
                                   int M, int N, int K, int act, hipStream_t s)
{
    int tiles = (M / 32) * (N / 32);
    int blocks = (tiles + 3) / 4;
    wmma_gemm<false><<<blocks, 128, 0, s>>>((const void*)A, lda, W, b, C, ldc, M, N, K, act);
}
static inline void launch_gemm_f16(const _Float16* A, long lda, const float* W,
                                   const float* b, float* C, long ldc,
                                   int M, int N, int K, int act, hipStream_t s)
{
    int tiles = (M / 32) * (N / 32);
    int blocks = (tiles + 3) / 4;
    wmma_gemm<true><<<blocks, 128, 0, s>>>((const void*)A, lda, W, b, C, ldc, M, N, K, act);
}
static inline void launch_ln(const float* src, const float* res, int rm,
                             const float* g, const float* b, float* dst,
                             int M, int F, int act, hipStream_t s)
{
    ln_kernel<<<M, 32, 0, s>>>(src, res, rm, g, b, dst, F, act);
}

extern "C" void kernel_launch(void* const* d_in, const int* in_sizes, int n_in,
                              void* d_out, int out_size, void* d_ws, size_t ws_size,
                              hipStream_t stream)
{
    (void)in_sizes; (void)n_in; (void)out_size; (void)ws_size;
    const int Bn = 512, T = 40, M = Bn * T;   // 20480 tokens

    const float* feat = (const float*)d_in[0];
    const float* tseq = (const float*)d_in[1];
    int p = 2;
    auto nx = [&]() { return (const float*)d_in[p++]; };
    const float *enc_in_w = nx(), *enc_in_b = nx(), *enc_ln_g = nx(), *enc_ln_b = nx();
    const float *r1l1w = nx(), *r1l1b = nx(), *r1n1g = nx(), *r1n1b = nx();
    const float *r1l2w = nx(), *r1l2b = nx(), *r1n2g = nx(), *r1n2b = nx();
    const float *r2l1w = nx(), *r2l1b = nx(), *r2n1g = nx(), *r2n1b = nx();
    const float *r2l2w = nx(), *r2l2b = nx(), *r2n2g = nx(), *r2n2b = nx();
    const float *eow = nx(), *eob = nx(), *eolg = nx(), *eolb = nx();
    const float *c1w = nx(), *c1b = nx(), *b1g = nx(), *b1b = nx(), *b1m = nx(), *b1v = nx();
    const float *c2w = nx(), *c2b = nx(), *b2g = nx(), *b2b = nx(), *b2m = nx(), *b2v = nx();
    const float *g0f_wih = nx(), *g0f_whh = nx(), *g0f_bih = nx(), *g0f_bhh = nx();
    const float *g0b_wih = nx(), *g0b_whh = nx(), *g0b_bih = nx(), *g0b_bhh = nx();
    const float *g1f_wih = nx(), *g1f_whh = nx(), *g1f_bih = nx(), *g1f_bhh = nx();
    const float *g1b_wih = nx(), *g1b_whh = nx(), *g1b_bih = nx(), *g1b_bhh = nx();
    const float *tpw = nx(), *tpb = nx(), *tplg = nx(), *tplb = nx();
    const float *aiw = nx(), *aib = nx();
    const float *aow = nx(), *aob = nx(), *alg = nx(), *alb = nx();
    const float *w01w = nx(), *w01b = nx(), *w02w = nx(), *w02b = nx();
    const float *la = nx(), *lb = nx(), *lc = nx(), *ld = nx();
    const float *s1w = nx(), *s1b = nx(), *s2w = nx(), *s2b = nx();
    const float *n1w = nx(), *n1b = nx(), *nlg = nx(), *nlb = nx();
    const float *n2w = nx(), *n2b = nx(), *n3w = nx(), *n3b = nx();
    const float *fl1w = nx(), *fl1b = nx(), *fl2w = nx(), *fl2b = nx();
    const float *ru1w = nx(), *ru1b = nx(), *ru2w = nx(), *ru2b = nx();
    const float *ph1w = nx(), *ph1b = nx(), *ph2w = nx(), *ph2b = nx();
    const float *un1w = nx(), *un1b = nx(), *un2w = nx(), *un2b = nx();
    const float *py1w = nx(), *py1b = nx(), *py2w = nx(), *py2b = nx();

    // ---- workspace layout (floats) ----
    float* Wsp = (float*)d_ws;
    size_t off = 0;
    auto alloc = [&](size_t n) { float* q = Wsp + off; off += n; return q; };
    float* X0 = alloc((size_t)M * 256);   // x / final z (ZF)
    float* T1 = alloc((size_t)M * 256);   // scratch
    float* T2 = alloc((size_t)M * 256);   // scratch / head outputs
    float* Z0 = alloc((size_t)M * 160);   // encoder z / z pre-attn
    float* RB = alloc((size_t)M * 768);   // im2col f16 | gru xp | qkv | zw(192)
    float* SM = alloc((size_t)2097152);
    float* hprev = SM;                     // 2*512*128
    float* hnew  = SM + 131072;            // 2*512*128
    float* hp    = SM + 262144;            // 2*512*384
    float* H512  = SM + 655360;            // 512*64
    float* tmean = SM + 688128;            // 64
    float* Wall  = SM + 688192;            // 20480
    float* WPAD  = SM + 708672;            // 4 * 64*192 padded head weights

    // ---- encoder ----
    launch_gemm_f32(feat, 128, enc_in_w, enc_in_b, T1, 256, M, 256, 128, ACT_NONE, stream);
    launch_ln(T1, nullptr, 0, enc_ln_g, enc_ln_b, X0, M, 256, ACT_GELU, stream);
    // res1
    launch_gemm_f32(X0, 256, r1l1w, r1l1b, T1, 256, M, 256, 256, ACT_NONE, stream);
    launch_ln(T1, nullptr, 0, r1n1g, r1n1b, T2, M, 256, ACT_GELU, stream);
    launch_gemm_f32(T2, 256, r1l2w, r1l2b, T1, 256, M, 256, 256, ACT_NONE, stream);
    launch_ln(T1, X0, 2, r1n2g, r1n2b, X0, M, 256, ACT_GELU, stream);
    // res2
    launch_gemm_f32(X0, 256, r2l1w, r2l1b, T1, 256, M, 256, 256, ACT_NONE, stream);
    launch_ln(T1, nullptr, 0, r2n1g, r2n1b, T2, M, 256, ACT_GELU, stream);
    launch_gemm_f32(T2, 256, r2l2w, r2l2b, T1, 256, M, 256, 256, ACT_NONE, stream);
    launch_ln(T1, X0, 2, r2n2g, r2n2b, X0, M, 256, ACT_GELU, stream);
    // enc_out
    launch_gemm_f32(X0, 256, eow, eob, T1, 160, M, 160, 256, ACT_NONE, stream);
    launch_ln(T1, nullptr, 0, eolg, eolb, Z0, M, 160, ACT_NONE, stream);

    // ---- conv residual block (im2col -> K=800 WMMA GEMMs) ----
    _Float16* COL = (_Float16*)RB;
    {
        long tot = (long)M * 800;
        im2col_k5<<<(int)((tot + 255) / 256), 256, 0, stream>>>(Z0, COL, Bn, T, 160);
        launch_gemm_f16(COL, 800, c1w, c1b, T1, 160, M, 160, 800, ACT_GELU, stream);
        bn_affine<<<(int)(((long)M * 160 + 255) / 256), 256, 0, stream>>>(T1, b1g, b1b, b1m, b1v, 160, (long)M * 160);
        im2col_k5<<<(int)((tot + 255) / 256), 256, 0, stream>>>(T1, COL, Bn, T, 160);
        launch_gemm_f16(COL, 800, c2w, c2b, T2, 160, M, 160, 800, ACT_GELU, stream);
        bn_affine<<<(int)(((long)M * 160 + 255) / 256), 256, 0, stream>>>(T2, b2g, b2b, b2m, b2v, 160, (long)M * 160);
        add_kernel<<<(int)(((long)M * 160 + 255) / 256), 256, 0, stream>>>(Z0, T2, X0, (long)M * 160);
    }

    // ---- bidirectional GRU (2 layers) ----
    float* xpf = RB;
    float* xpb = RB + (size_t)M * 384;
    for (int layer = 0; layer < 2; layer++) {
        const float* in  = (layer == 0) ? X0 : T1;
        int din          = (layer == 0) ? 160 : 256;
        float* ys        = (layer == 0) ? T1 : T2;
        const float *wihF = (layer == 0) ? g0f_wih : g1f_wih;
        const float *bihF = (layer == 0) ? g0f_bih : g1f_bih;
        const float *whhF = (layer == 0) ? g0f_whh : g1f_whh;
        const float *bhhF = (layer == 0) ? g0f_bhh : g1f_bhh;
        const float *wihB = (layer == 0) ? g0b_wih : g1b_wih;
        const float *bihB = (layer == 0) ? g0b_bih : g1b_bih;
        const float *whhB = (layer == 0) ? g0b_whh : g1b_whh;
        const float *bhhB = (layer == 0) ? g0b_bhh : g1b_bhh;
        launch_gemm_f32(in, din, wihF, bihF, xpf, 384, M, 384, din, ACT_NONE, stream);
        launch_gemm_f32(in, din, wihB, bihB, xpb, 384, M, 384, din, ACT_NONE, stream);
        zero_kernel<<<(131072 + 255) / 256, 256, 0, stream>>>(hprev, 131072);
        float* hc = hprev; float* hx = hnew;
        for (int s = 0; s < T; s++) {
            launch_gemm_f32(hc,         128, whhF, bhhF, hp,             384, 512, 384, 128, ACT_NONE, stream);
            launch_gemm_f32(hc + 65536, 128, whhB, bhhB, hp + 512 * 384, 384, 512, 384, 128, ACT_NONE, stream);
            dim3 g((512 * 128 + 255) / 256, 2);
            gru_gate<<<g, 256, 0, stream>>>(xpf, xpb, hp, hp + 512 * 384,
                                            hc, hc + 65536, hx, hx + 65536,
                                            ys, s, T, 256);
            float* t = hc; hc = hx; hx = t;
        }
    }

    // ---- tproj + attention ----
    launch_gemm_f32(T2, 256, tpw, tpb, T1, 160, M, 160, 256, ACT_NONE, stream);
    launch_ln(T1, nullptr, 0, tplg, tplb, Z0, M, 160, ACT_GELU, stream);   // Z0 = z pre-attn
    launch_gemm_f32(Z0, 160, aiw, aib, RB, 480, M, 480, 160, ACT_NONE, stream);
    attn_kernel<<<Bn * 4, 256, 0, stream>>>(RB, T1);
    launch_gemm_f32(T1, 160, aow, aob, T2, 160, M, 160, 160, ACT_NONE, stream);
    launch_ln(T2, Z0, 1, alg, alb, X0, M, 160, ACT_NONE, stream);          // X0 = final z (ZF)
    float* ZF = X0;

    // ---- W0 head + tmean ----
    tmean_kernel<<<1, 64, 0, stream>>>(tseq, tmean);
    launch_gemm_f32(ZF, (long)T * 160, w01w, w01b, H512, 64, 512, 64, 160, ACT_GELU, stream);
    head_out<<<(512 + 3) / 4, 128, 0, stream>>>(H512, w02w, w02b, Wall, 40, 1, ACT_SIGMOID, 512);

    // ---- neural ODE ----
    ode_kernel<<<Bn, 160, 0, stream>>>(ZF, tmean, la, lb, lc, ld,
                                       s1w, s1b, s2w, s2b, n1w, n1b, nlg, nlb,
                                       n2w, n2b, n3w, n3b, Wall);

    // ---- heads (zw padded to K=192; head layer-1 weights padded to match) ----
    float* ZW = RB;  // [M,192]
    zw_build<<<(int)(((long)M * 192 + 255) / 256), 256, 0, stream>>>(ZF, Wall, ZW, M);
    float* FLW = WPAD;
    float* RUW = WPAD + 12288;
    float* PHW = WPAD + 24576;
    float* UNW = WPAD + 36864;
    pad_w64<<<(64 * 192 + 255) / 256, 256, 0, stream>>>(fl1w, FLW);
    pad_w64<<<(64 * 192 + 255) / 256, 256, 0, stream>>>(ru1w, RUW);
    pad_w64<<<(64 * 192 + 255) / 256, 256, 0, stream>>>(ph1w, PHW);
    pad_w64<<<(64 * 192 + 255) / 256, 256, 0, stream>>>(un1w, UNW);
    float* FL = T2;
    float* RU = T2 + (size_t)M * 3;
    float* PH = T2 + (size_t)M * 4;
    float* UN = T2 + (size_t)M * 7;
    float* PY = T2 + (size_t)M * 8;
    launch_gemm_f32(ZW, 192, FLW, fl1b, T1, 64, M, 64, 192, ACT_GELU, stream);
    head_out<<<(M + 3) / 4, 128, 0, stream>>>(T1, fl2w, fl2b, FL, 3, 3, ACT_NONE, M);
    launch_gemm_f32(ZW, 192, RUW, ru1b, T1, 64, M, 64, 192, ACT_GELU, stream);
    head_out<<<(M + 3) / 4, 128, 0, stream>>>(T1, ru2w, ru2b, RU, 1, 1, ACT_SIGMOID, M);
    launch_gemm_f32(ZW, 192, PHW, ph1b, T1, 64, M, 64, 192, ACT_GELU, stream);
    head_out<<<(M + 3) / 4, 128, 0, stream>>>(T1, ph2w, ph2b, PH, 3, 3, ACT_NONE, M);
    launch_gemm_f32(ZW, 192, UNW, un1b, T1, 64, M, 64, 192, ACT_GELU, stream);
    head_out<<<(M + 3) / 4, 128, 0, stream>>>(T1, un2w, un2b, UN, 1, 1, ACT_NONE, M);
    launch_gemm_f32(ZF, 160, py1w, py1b, T1, 64, M, 64, 160, ACT_GELU, stream);
    head_out<<<(M + 3) / 4, 128, 0, stream>>>(T1, py2w, py2b, PY, 2, 2, ACT_NONE, M);

    // ---- assemble [B,T,171] ----
    assemble<<<M, 192, 0, stream>>>(ZF, Wall, FL, RU, PH, UN, PY, (float*)d_out);
}
